// MultiHeadAttention_11175504904663
// MI455X (gfx1250) — matrix operand order repaired and checked by reading, not compiled, not run on hardware
//
#include <hip/hip_runtime.h>
#include <hip/hip_bf16.h>

// ---------------------------------------------------------------------------
// MultiHeadAttention w/ relative bias for MI455X (gfx1250, wave32, WMMA).
//
// Shapes: B=8, S=512 (Q=K), D=1024, H=16, dk=64, SCALE=8.
// Algebraic refactor of the rel-bias term (saves ~480 GFLOP + 1GB scratch):
//   rel[b,h,q,k] = (1/8) * sum_D rb[q,k,D] * U[b,h,q,D] + (1/8)*dot(br_h,Qh)
//   U[b,h,q,D]   = sum_d Qh[b,h,q,d] * Wr[D, h*64+d]
// All GEMMs use v_wmma_f32_16x16x32_bf16 (bf16 in, f32 accum); bf16 tiles are
// staged with GLOBAL_LOAD_ASYNC_TO_LDS_B128 (double-buffered, ASYNCcnt),
// fp32 tiles go through VGPR + v_cvt_pk_bf16_f32.
//
// Workspace layout: needs ~178 MB (see offsets in kernel_launch).
// d_out: out fp32 (8,512,1024) then attn fp32 (8,16,512,512).
// ---------------------------------------------------------------------------

typedef __bf16 bf16;
typedef __attribute__((ext_vector_type(16))) __bf16 v16bf;
typedef __attribute__((ext_vector_type(8)))  __bf16 bf16x8;
typedef __attribute__((ext_vector_type(8)))  float  v8f;
typedef __attribute__((ext_vector_type(4)))  float  f32x4;

union FragAB { bf16x8 h[2]; v16bf v; };

constexpr int TM = 128, TN = 128, TK = 32;
constexpr int LDSTR = TK + 8;          // 40 bf16 = 80B row stride (16B aligned)

#ifndef __has_builtin
#define __has_builtin(x) 0
#endif

#if __has_builtin(__builtin_amdgcn_global_load_async_to_lds_b128)
#define ASYNC_LDS 1
// Builtin signature (from clang diagnostic): arg0 = int4 addrspace(1)*,
// arg1 = int4 addrspace(3)*, arg2 = imm offset, arg3 = imm cpol.
typedef __attribute__((ext_vector_type(4))) int i32x4;
typedef __attribute__((address_space(1))) i32x4 ga_i32x4;
typedef __attribute__((address_space(3))) i32x4 la_i32x4;
#else
#define ASYNC_LDS 0
#endif

__device__ inline void wait_async() {
#if ASYNC_LDS
#if __has_builtin(__builtin_amdgcn_s_wait_asynccnt)
  __builtin_amdgcn_s_wait_asynccnt(0);
#else
  asm volatile("s_wait_asynccnt 0" ::: "memory");
#endif
#endif
}

struct GemmP {
  const void* A;  const void* Bt;  void* C;  const float* bias;
  long a_base, b_base, c_base;
  long a_rs, b_rs;          // row strides in elements: A[r][k], Bt[n][k]
  long c_rs, c_cs;          // C addressing: c_base + row*c_rs + col*c_cs
  long a_z1, a_z2, b_z1, b_z2, c_z1, c_z2;  // z1 = z/zmod, z2 = z%zmod
  int  zmod;
  int  M, N, K;             // K must be a multiple of 32
  int  bias_off;
  float scale;
};

// Stage a 128x32 tile into LDS as bf16.  256 threads: thread t handles row
// t>>1, 16-elem segment (t&1)*16.  bf16 sources use async global->LDS DMA;
// fp32 sources are loaded to VGPRs and packed with v_cvt_pk_bf16_f32.
template <typename T>
__device__ inline void stage(bf16* sm, const T* g, long base, long rs,
                             int rows, int k0, int tid) {
  int r  = tid >> 1;
  int cs = (tid & 1) << 4;
  bf16* o = sm + r * LDSTR + cs;
  if (r < rows) {
    const T* p = g + base + (long)r * rs + k0 + cs;
    if constexpr (sizeof(T) == 2) {
#if ASYNC_LDS
      // 32B per thread = two b128 async copies (imm offset applies to both
      // the global and the LDS address per the ISA's async addressing).
      __builtin_amdgcn_global_load_async_to_lds_b128(
          (ga_i32x4*)(void*)p, (la_i32x4*)(void*)o, 0, 0);
      __builtin_amdgcn_global_load_async_to_lds_b128(
          (ga_i32x4*)(void*)p, (la_i32x4*)(void*)o, 16, 0);
#else
      ((bf16x8*)o)[0] = ((const bf16x8*)p)[0];
      ((bf16x8*)(o + 8))[0] = ((const bf16x8*)p)[1];
#endif
    } else {
      const f32x4* pv = (const f32x4*)p;       // 16B-aligned by construction
      __builtin_prefetch((const void*)(p + TK), 0, 1);   // next K-tile hint
#pragma unroll
      for (int j = 0; j < 4; ++j) {
        f32x4 v = pv[j];
        o[j*4+0] = (bf16)v.x; o[j*4+1] = (bf16)v.y;
        o[j*4+2] = (bf16)v.z; o[j*4+3] = (bf16)v.w;
      }
    }
  } else {
#pragma unroll
    for (int j = 0; j < 16; ++j) o[j] = (bf16)0.0f;
  }
}

// Generic WMMA GEMM: C = scale*(A @ Bt^T) [+ bias][+= old C].
// Workgroup tile 128x128, 8 waves (2 x 4), wave tile 64x32 = 4x2 WMMA tiles.
// Double-buffered K-loop: tile i+1 is staged (async) while tile i computes.
template <typename AT, typename BT, typename OT, bool ACCUM, bool BIAS>
__global__ __launch_bounds__(256) void wmma_gemm(GemmP p) {
  __shared__ bf16 smem[2][(TM + TN) * LDSTR];      // 2 x 20KB
  constexpr bool USE_ASYNC = ASYNC_LDS && (sizeof(AT) == 2 || sizeof(BT) == 2);

  const int tid  = threadIdx.x;
  const int lane = tid & 31;
  const int wid  = tid >> 5;
  const int wm   = wid & 1;       // 2 wave-rows of 64
  const int wn   = wid >> 1;      // 4 wave-cols of 32
  const int bm   = blockIdx.x, bn = blockIdx.y;
  const int z    = blockIdx.z;
  const int z2   = z % p.zmod;
  const int z1   = z / p.zmod;

  const long aoff = p.a_base + (long)z1 * p.a_z1 + (long)z2 * p.a_z2
                  + (long)bm * TM * p.a_rs;
  const long boff = p.b_base + (long)z1 * p.b_z1 + (long)z2 * p.b_z2
                  + (long)bn * TN * p.b_rs;
  const long coff = p.c_base + (long)z1 * p.c_z1 + (long)z2 * p.c_z2;
  const int arows = p.M - bm * TM;
  const int brows = p.N - bn * TN;

  const AT* A = (const AT*)p.A;
  const BT* B = (const BT*)p.Bt;
  OT*       C = (OT*)p.C;

  v8f acc[4][2];
#pragma unroll
  for (int i = 0; i < 4; ++i)
#pragma unroll
    for (int j = 0; j < 2; ++j)
#pragma unroll
      for (int r = 0; r < 8; ++r) acc[i][j][r] = 0.0f;

  const int half = lane >> 4;     // 0/1: lane group
  const int l15  = lane & 15;

  auto stageAB = [&](int buf, int k0) {
    stage<AT>(smem[buf],              A, aoff, p.a_rs, arows, k0, tid);
    stage<BT>(smem[buf] + TM * LDSTR, B, boff, p.b_rs, brows, k0, tid);
  };

  stageAB(0, 0);
  if constexpr (USE_ASYNC) wait_async();
  __syncthreads();

  int cur = 0;
  for (int k0 = 0; k0 < p.K; k0 += TK) {
    const bool more = (k0 + TK < p.K);
    if (more) stageAB(cur ^ 1, k0 + TK);     // overlap with WMMA below

    const bf16* sA = smem[cur];
    const bf16* sB = smem[cur] + TM * LDSTR;

    FragAB a[4], b[2];
#pragma unroll
    for (int mt = 0; mt < 4; ++mt) {
      // 16-bit A 16x32 layout: lane<16 holds K 0..7 & 16..23; lane>=16 holds
      // K 8..15 & 24..31 (packed pairs per VGPR).
      const bf16* ap = sA + (wm * 64 + mt * 16 + l15) * LDSTR + half * 8;
      a[mt].h[0] = *(const bf16x8*)ap;
      a[mt].h[1] = *(const bf16x8*)(ap + 16);
    }
#pragma unroll
    for (int nt = 0; nt < 2; ++nt) {
      // B 32x16: lane = column n; Bt[n][k] rows in LDS; same K split as A:
      // lane<16 holds K 0..7 & 16..23; lane>=16 holds K 8..15 & 24..31.
      const bf16* bp = sB + (wn * 32 + nt * 16 + l15) * LDSTR + half * 8;
      b[nt].h[0] = *(const bf16x8*)bp;
      b[nt].h[1] = *(const bf16x8*)(bp + 16);
    }
#pragma unroll
    for (int mt = 0; mt < 4; ++mt)
#pragma unroll
      for (int nt = 0; nt < 2; ++nt)
        acc[mt][nt] = __builtin_amdgcn_wmma_f32_16x16x32_bf16(
            false, a[mt].v, false, b[nt].v, (short)0, acc[mt][nt], false, false);

    if constexpr (USE_ASYNC) { if (more) wait_async(); }
    __syncthreads();
    cur ^= 1;
  }

  // Epilogue per 16x16 f32 C layout: c[r] at row 8*half + r, col = l15.
#pragma unroll
  for (int mt = 0; mt < 4; ++mt) {
#pragma unroll
    for (int nt = 0; nt < 2; ++nt) {
      const int colg = bn * TN + wn * 32 + nt * 16 + l15;
      if (colg >= p.N) continue;
      float bv = 0.0f;
      if (BIAS) bv = p.bias[p.bias_off + colg];
#pragma unroll
      for (int r = 0; r < 8; ++r) {
        const int rowg = bm * TM + wm * 64 + mt * 16 + half * 8 + r;
        if (rowg >= p.M) continue;
        const long off = coff + (long)rowg * p.c_rs + (long)colg * p.c_cs;
        float v = acc[mt][nt][r] * p.scale + bv;
        if (ACCUM) v += (float)C[off];
        C[off] = (OT)v;
      }
    }
  }
}

// fp32 1024x1024 weight -> bf16, optionally transposed into Bt[n][k] layout.
__global__ __launch_bounds__(256) void convert_weight(const float* W, bf16* out, int trans) {
  const int idx = blockIdx.x * 256 + threadIdx.x;   // n*1024 + k
  const int n = idx >> 10, k = idx & 1023;
  const float v = trans ? W[k * 1024 + n] : W[n * 1024 + k];
  out[idx] = (bf16)v;
}

// Per-batch bf16 transpose: in (R x C) -> out (C x R).  Block (32,8).
__global__ void transpose_bf16(const bf16* in, bf16* out, int R, int C) {
  __shared__ bf16 tile[32][33];
  const long base = (long)blockIdx.z * (long)R * (long)C;
  const int r0 = blockIdx.y * 32, c0 = blockIdx.x * 32;
  const int tx = threadIdx.x, ty = threadIdx.y;
#pragma unroll
  for (int i = 0; i < 32; i += 8)
    tile[ty + i][tx] = in[base + (long)(r0 + ty + i) * C + (c0 + tx)];
  __syncthreads();
#pragma unroll
  for (int i = 0; i < 32; i += 8)
    out[base + (long)(c0 + ty + i) * R + (r0 + tx)] = tile[tx][ty + i];
}

// Row-wise: add (1/8)*dot(br_h, Qh_h) (the folded br term), apply mask==1 ->
// -1e10, softmax over the 512 keys.  One 128-thread block per (b,h,q) row.
__global__ __launch_bounds__(128) void softmax_kernel(float* attn, const int* mask,
                                                      const bf16* Qh, const float* br) {
  __shared__ float red[128];
  const int row = blockIdx.x;           // (b*16+h)*512 + q
  const int q  = row & 511;
  const int bh = row >> 9;
  const int h  = bh & 15;
  const int b  = bh >> 4;
  float* s = attn + (long)row * 512;
  const int* m = mask + (long)b * 512;
  const int t = threadIdx.x;

  float part = 0.0f;
  if (t < 64)
    part = br[h * 64 + t] * (float)Qh[((long)(b * 512 + q)) * 1024 + h * 64 + t];
  red[t] = part;
  __syncthreads();
  for (int w = 64; w > 0; w >>= 1) { if (t < w) red[t] += red[t + w]; __syncthreads(); }
  const float cb = red[0] * 0.125f;
  __syncthreads();

  float v[4];
  float mx = -3.0e38f;
#pragma unroll
  for (int j = 0; j < 4; ++j) {
    const int k = t + j * 128;
    float x = s[k] + cb;
    if (m[k] == 1) x = -1.0e10f;
    v[j] = x;
    mx = fmaxf(mx, x);
  }
  red[t] = mx; __syncthreads();
  for (int w = 64; w > 0; w >>= 1) { if (t < w) red[t] = fmaxf(red[t], red[t + w]); __syncthreads(); }
  mx = red[0]; __syncthreads();

  float sum = 0.0f;
#pragma unroll
  for (int j = 0; j < 4; ++j) { v[j] = __expf(v[j] - mx); sum += v[j]; }
  red[t] = sum; __syncthreads();
  for (int w = 64; w > 0; w >>= 1) { if (t < w) red[t] += red[t + w]; __syncthreads(); }
  const float inv = 1.0f / red[0];
#pragma unroll
  for (int j = 0; j < 4; ++j) s[t + j * 128] = v[j] * inv;
}

extern "C" void kernel_launch(void* const* d_in, const int* in_sizes, int n_in,
                              void* d_out, int out_size, void* d_ws, size_t ws_size,
                              hipStream_t stream) {
  const float* query = (const float*)d_in[0];
  const float* key   = (const float*)d_in[1];
  const float* value = (const float*)d_in[2];
  const int*   mask  = (const int*)d_in[3];
  const float* rb    = (const float*)d_in[4];
  const float* Wq = (const float*)d_in[5];  const float* bq = (const float*)d_in[6];
  const float* Wk = (const float*)d_in[7];  const float* bk = (const float*)d_in[8];
  const float* Wv = (const float*)d_in[9];  const float* bv = (const float*)d_in[10];
  const float* Wr = (const float*)d_in[11]; const float* br = (const float*)d_in[12];
  const float* Wo = (const float*)d_in[13]; const float* bo = (const float*)d_in[14];

  float* out  = (float*)d_out;
  float* attn = out + 8L * 512 * 1024;       // 4,194,304 floats of output first

  char* ws = (char*)d_ws;
  const size_t MB = 1u << 20;
  bf16* WqT = (bf16*)(ws +  0 * MB);
  bf16* WkT = (bf16*)(ws +  2 * MB);
  bf16* WvT = (bf16*)(ws +  4 * MB);
  bf16* WoT = (bf16*)(ws +  6 * MB);
  bf16* WrC = (bf16*)(ws +  8 * MB);
  bf16* Qh  = (bf16*)(ws + 10 * MB);
  bf16* Kh  = (bf16*)(ws + 18 * MB);
  bf16* Vh  = (bf16*)(ws + 26 * MB);
  bf16* VhT = (bf16*)(ws + 34 * MB);
  bf16* ctx = (bf16*)(ws + 42 * MB);
  bf16* U   = (bf16*)(ws + 50 * MB);         // 128 MB: (B,H,S,1024)

  // --- 1. weight conversion (transposed into Bt[n][k]) --------------------
  convert_weight<<<4096, 256, 0, stream>>>(Wq, WqT, 1);
  convert_weight<<<4096, 256, 0, stream>>>(Wk, WkT, 1);
  convert_weight<<<4096, 256, 0, stream>>>(Wv, WvT, 1);
  convert_weight<<<4096, 256, 0, stream>>>(Wo, WoT, 1);
  convert_weight<<<4096, 256, 0, stream>>>(Wr, WrC, 0);   // U wants Wr as-is

  // --- 2. Q/K/V projections: (4096x1024)@(1024x1024)+bias -> bf16 ---------
  GemmP p{};
  p.a_rs = 1024; p.b_rs = 1024; p.c_rs = 1024; p.c_cs = 1;
  p.zmod = 1; p.M = 4096; p.N = 1024; p.K = 1024; p.scale = 1.0f;
  dim3 gproj(32, 8, 1);
  p.A = query; p.Bt = WqT; p.C = Qh; p.bias = bq;
  wmma_gemm<float, bf16, bf16, false, true><<<gproj, 256, 0, stream>>>(p);
  p.A = key;   p.Bt = WkT; p.C = Kh; p.bias = bk;
  wmma_gemm<float, bf16, bf16, false, true><<<gproj, 256, 0, stream>>>(p);
  p.A = value; p.Bt = WvT; p.C = Vh; p.bias = bv;
  wmma_gemm<float, bf16, bf16, false, true><<<gproj, 256, 0, stream>>>(p);

  // --- 3. Vh -> VhT per batch (for attn @ V) -------------------------------
  transpose_bf16<<<dim3(32, 16, 8), dim3(32, 8), 0, stream>>>(Vh, VhT, 512, 1024);

  // --- 4. U[b,h,q,D] = Qh_h @ Wr_h^T  (z = b*16+h, M=512,N=1024,K=64) ------
  p = GemmP{};
  p.A = Qh; p.Bt = WrC; p.C = U; p.bias = nullptr;
  p.a_rs = 1024; p.a_z1 = 524288; p.a_z2 = 64;
  p.b_rs = 1024; p.b_z1 = 0;      p.b_z2 = 64;
  p.c_rs = 1024; p.c_cs = 1; p.c_z1 = 8388608; p.c_z2 = 524288;
  p.zmod = 16; p.M = 512; p.N = 1024; p.K = 64; p.scale = 1.0f;
  wmma_gemm<bf16, bf16, bf16, false, false><<<dim3(4, 8, 128), 256, 0, stream>>>(p);

  // --- 5. scores = Qh @ Kh^T / 8 -> attn region (z = b*16+h) ---------------
  p = GemmP{};
  p.A = Qh; p.Bt = Kh; p.C = attn; p.bias = nullptr;
  p.a_rs = 1024; p.a_z1 = 524288; p.a_z2 = 64;
  p.b_rs = 1024; p.b_z1 = 524288; p.b_z2 = 64;
  p.c_rs = 512; p.c_cs = 1; p.c_z1 = 4194304; p.c_z2 = 262144;
  p.zmod = 16; p.M = 512; p.N = 512; p.K = 64; p.scale = 0.125f;
  wmma_gemm<bf16, bf16, float, false, false><<<dim3(4, 4, 128), 256, 0, stream>>>(p);

  // --- 6. rel: per q, scores[bh, q, kpos] += (1/8) U[bh,q,:] . rb[q,kpos,:]
  //     M=128 (bh), N=512 (kpos), K=1024 (D); rb streamed once total. -------
  p = GemmP{};
  p.A = U; p.Bt = rb; p.C = attn; p.bias = nullptr;
  p.a_rs = 524288; p.a_z1 = 1024;
  p.b_rs = 1024;   p.b_z1 = 524288;
  p.c_rs = 262144; p.c_cs = 1; p.c_z1 = 512;
  p.zmod = 1; p.M = 128; p.N = 512; p.K = 1024; p.scale = 0.125f;
  wmma_gemm<bf16, float, float, true, false><<<dim3(1, 4, 512), 256, 0, stream>>>(p);

  // --- 7. + br-dot, mask, softmax ------------------------------------------
  softmax_kernel<<<65536, 128, 0, stream>>>(attn, mask, Qh, br);

  // --- 8. context = attn @ Vh (z = b*16+h, M=512,N=64,K=512) ---------------
  p = GemmP{};
  p.A = attn; p.Bt = VhT; p.C = ctx; p.bias = nullptr;
  p.a_rs = 512; p.a_z1 = 4194304; p.a_z2 = 262144;
  p.b_rs = 512; p.b_z1 = 524288;  p.b_z2 = 32768;
  p.c_rs = 1024; p.c_cs = 1; p.c_z1 = 524288; p.c_z2 = 64;
  p.zmod = 16; p.M = 512; p.N = 64; p.K = 512; p.scale = 1.0f;
  wmma_gemm<float, bf16, bf16, false, false><<<dim3(4, 1, 128), 256, 0, stream>>>(p);

  // --- 9. output = ctx @ Wo + bo -> fp32 -----------------------------------
  p = GemmP{};
  p.A = ctx; p.Bt = WoT; p.C = out; p.bias = bo;
  p.a_rs = 1024; p.b_rs = 1024; p.c_rs = 1024; p.c_cs = 1;
  p.zmod = 1; p.M = 4096; p.N = 1024; p.K = 1024; p.scale = 1.0f;
  wmma_gemm<bf16, bf16, float, false, true><<<dim3(32, 8, 1), 256, 0, stream>>>(p);
}
